// NodeEncoder_45526653337885
// MI455X (gfx1250) — matrix operand order repaired and checked by reading, not compiled, not run on hardware
//
#include <hip/hip_runtime.h>
#include <hip/hip_bf16.h>
#include <stdint.h>

typedef __attribute__((ext_vector_type(16))) _Float16 v16h;
typedef __attribute__((ext_vector_type(8)))  _Float16 v8h;
typedef __attribute__((ext_vector_type(8)))  float    v8f;

#define ROI_  7
#define CIN   256
#define HW_   40
#define NROI  4096          // N*K = 512*8
#define K1    12544         // 256*49
#define HID   512
#define DOUT  512
#define EPSV  0.001f

// ------------------------------------------------------------------
// CDNA5 async global->LDS copy (ASYNCcnt-tracked), 16B per lane.
// vdst = LDS byte address (low 32 bits of generic pointer), vaddr = global.
// ------------------------------------------------------------------
__device__ __forceinline__ void async_ld_b128(uint32_t lds_addr, const void* gaddr) {
  asm volatile("global_load_async_to_lds_b128 %0, %1, off"
               :: "v"(lds_addr), "v"(gaddr) : "memory");
}
__device__ __forceinline__ void wait_async0() {
  asm volatile("s_wait_asynccnt 0x0" ::: "memory");
}
__device__ __forceinline__ uint32_t lds_addr_of(const void* p) {
  return (uint32_t)(uintptr_t)p;   // LDS aperture: low 32 bits = LDS byte offset
}

// ------------------------------------------------------------------
// RoIAlign (sampling_ratio=1, aligned=False) -> A matrix [NROI, K1] f16
// k = c*49 + h*7 + w  (matches w1 'ochw' flattening)
// ------------------------------------------------------------------
__global__ void roi_align_f16(const float* __restrict__ images,
                              const int*   __restrict__ kfidx,
                              const float* __restrict__ boxes,
                              _Float16*    __restrict__ Aout) {
  const int r = blockIdx.x;
  const int f = kfidx[r];
  const float bx1 = boxes[r * 4 + 0] * (float)HW_;
  const float by1 = boxes[r * 4 + 1] * (float)HW_;
  const float bx2 = boxes[r * 4 + 2] * (float)HW_;
  const float by2 = boxes[r * 4 + 3] * (float)HW_;
  const float bw = (bx2 - bx1) * (1.0f / ROI_);
  const float bh = (by2 - by1) * (1.0f / ROI_);
  const float* __restrict__ img = images + (size_t)f * CIN * HW_ * HW_;

  for (int k = threadIdx.x; k < K1; k += blockDim.x) {
    int c  = k / 49;
    int p  = k - c * 49;
    int ph = p / 7;
    int pw = p - ph * 7;
    float X = bx1 + ((float)pw + 0.5f) * bw;
    float Y = by1 + ((float)ph + 0.5f) * bh;
    bool valid = (X > -1.0f) && (X < (float)HW_) && (Y > -1.0f) && (Y < (float)HW_);
    float x = fminf(fmaxf(X, 0.0f), (float)(HW_ - 1));
    float y = fminf(fmaxf(Y, 0.0f), (float)(HW_ - 1));
    int x0 = (int)floorf(x), y0 = (int)floorf(y);
    int x1 = min(x0 + 1, HW_ - 1), y1 = min(y0 + 1, HW_ - 1);
    float lx = x - (float)x0, ly = y - (float)y0;
    const float* ch = img + (size_t)c * HW_ * HW_;
    float v00 = ch[y0 * HW_ + x0];
    float v01 = ch[y0 * HW_ + x1];
    float v10 = ch[y1 * HW_ + x0];
    float v11 = ch[y1 * HW_ + x1];
    float v = v00 * (1.0f - ly) * (1.0f - lx) + v01 * (1.0f - ly) * lx +
              v10 * ly * (1.0f - lx)          + v11 * ly * lx;
    Aout[(size_t)r * K1 + k] = (_Float16)(valid ? v : 0.0f);
  }
}

__global__ void cvt_f32_f16(const float* __restrict__ in,
                            _Float16* __restrict__ out, int n) {
  int i = blockIdx.x * blockDim.x + threadIdx.x;
  if (i < n) out[i] = (_Float16)in[i];
}

// ------------------------------------------------------------------
// Fused GEMM + BN + ReLU (+ optional max-over-8-rows).
// A: [M,K] f16 row-major, B: [N,K] f16 row-major (k contiguous).
// Tile 128x64x32, 128 threads (4 waves). Wave w owns M rows
// [w*32, w*32+32): 2 M-subtiles x 4 N-subtiles = 8 WMMA accumulators.
// Double-buffered LDS fed by async global->LDS DMA; one barrier/k-step.
// MODE 0: store f16 y -> [M, HID]
// MODE 1: store f32 max over each aligned group of 8 rows -> [M/8, DOUT]
//         (C/D vgpr j holds rows base+hi*8+j: frame-max is in-register)
// ------------------------------------------------------------------
union AF16 { v16h v; v8h h[2]; };

template <int MODE>
__launch_bounds__(128, 1)
__global__ void gemm_bn_relu(const _Float16* __restrict__ A,
                             const _Float16* __restrict__ Bm,
                             int K,
                             const float* __restrict__ bias,
                             const float* __restrict__ gamma,
                             const float* __restrict__ beta,
                             const float* __restrict__ mean,
                             const float* __restrict__ var,
                             void* __restrict__ outp) {
  constexpr int BM = 128, BN = 64, BK = 32;
  constexpr int LDSW = BK + 8;  // 40 halfs = 80 B rows: 16B-aligned, conflict-padded
  __shared__ __align__(16) _Float16 As[2][BM][LDSW];
  __shared__ __align__(16) _Float16 Bs[2][BN][LDSW];

  const int tid  = threadIdx.x;        // 0..127
  const int wave = tid >> 5;           // 0..3
  const int lane = tid & 31;
  const int l16  = lane & 15;
  const int hi   = (lane >> 4) & 1;    // lane half-select

  const int m0 = blockIdx.x * BM;
  const int n0 = blockIdx.y * BN;

  // cooperative loaders: thread t owns A row t (32 halfs = 4x16B)
  // and B row t/2, 16-half chunk t%2 (1x... 2x8 halfs = 1x16B per chunk half)
  const _Float16* aptr = A  + (size_t)(m0 + tid) * K;
  const _Float16* bptr = Bm + (size_t)(n0 + (tid >> 1)) * K + (tid & 1) * 16;
  const int brow = tid >> 1, bcol = (tid & 1) * 16;

  // per-thread LDS destinations for the async DMA (two buffers)
  uint32_t aLds[2], bLds[2];
  aLds[0] = lds_addr_of(&As[0][tid][0]);
  aLds[1] = lds_addr_of(&As[1][tid][0]);
  bLds[0] = lds_addr_of(&Bs[0][brow][bcol]);
  bLds[1] = lds_addr_of(&Bs[1][brow][bcol]);

  auto issue_tile = [&](int it, int buf) {
    const int k0 = it * BK;
    async_ld_b128(aLds[buf],      aptr + k0);
    async_ld_b128(aLds[buf] + 16, aptr + k0 + 8);
    async_ld_b128(aLds[buf] + 32, aptr + k0 + 16);
    async_ld_b128(aLds[buf] + 48, aptr + k0 + 24);
    async_ld_b128(bLds[buf],      bptr + k0);
    async_ld_b128(bLds[buf] + 16, bptr + k0 + 8);
  };

  v8f acc[2][4] = {};

  const int nIter = K / BK;
  issue_tile(0, 0);                        // prologue: DMA tile 0

  for (int it = 0; it < nIter; ++it) {
    const int buf = it & 1;
    wait_async0();                         // this tile's DMA landed (mine)
    __syncthreads();                       // everyone's landed; prev reads done
    if (it + 1 < nIter)
      issue_tile(it + 1, buf ^ 1);         // DMA next tile under this compute

    // A fragments (16-bit A 16x32: lanes<16 K0-7/K16-23, lanes>=16 K8-15/K24-31)
    AF16 af[2];
#pragma unroll
    for (int mt = 0; mt < 2; ++mt) {
      const int am = wave * 32 + mt * 16 + l16;
      af[mt].h[0] = *(const v8h*)&As[buf][am][hi * 8];
      af[mt].h[1] = *(const v8h*)&As[buf][am][16 + hi * 8];
    }
    // B fragments (B 32x16: lanes<16 K0-7/K16-23, lanes>=16 K8-15/K24-31, N = lane%16)
    AF16 bf[4];
#pragma unroll
    for (int nt = 0; nt < 4; ++nt) {
      const int bn = nt * 16 + l16;
      bf[nt].h[0] = *(const v8h*)&Bs[buf][bn][hi * 8];
      bf[nt].h[1] = *(const v8h*)&Bs[buf][bn][16 + hi * 8];
    }
    // 8 independent WMMAs back-to-back
#pragma unroll
    for (int mt = 0; mt < 2; ++mt)
#pragma unroll
      for (int nt = 0; nt < 4; ++nt)
        acc[mt][nt] = __builtin_amdgcn_wmma_f32_16x16x32_f16(
            false, af[mt].v, false, bf[nt].v, (short)0, acc[mt][nt], false, false);
  }

  // Epilogue: BN affine + ReLU (+ frame max for MODE 1).
#pragma unroll
  for (int nt = 0; nt < 4; ++nt) {
    const int n = n0 + nt * 16 + l16;
    const float s  = gamma[n] * rsqrtf(var[n] + EPSV);
    const float sh = (bias[n] - mean[n]) * s + beta[n];
#pragma unroll
    for (int mt = 0; mt < 2; ++mt) {
      const int mbase = m0 + wave * 32 + mt * 16 + hi * 8;
      if (MODE == 0) {
#pragma unroll
        for (int j = 0; j < 8; ++j) {
          float v = fmaxf(acc[mt][nt][j] * s + sh, 0.0f);
          ((_Float16*)outp)[(size_t)(mbase + j) * HID + n] = (_Float16)v;
        }
      } else {
        float mx = acc[mt][nt][0];
#pragma unroll
        for (int j = 1; j < 8; ++j) mx = fmaxf(mx, acc[mt][nt][j]);
        const int node = mbase >> 3;  // 8 consecutive rows = one node's frames
        ((float*)outp)[(size_t)node * DOUT + n] = fmaxf(mx * s + sh, 0.0f);
      }
    }
  }
}

// ------------------------------------------------------------------
extern "C" void kernel_launch(void* const* d_in, const int* in_sizes, int n_in,
                              void* d_out, int out_size, void* d_ws, size_t ws_size,
                              hipStream_t stream) {
  const float* images = (const float*)d_in[0];
  const int*   kfidx  = (const int*)  d_in[1];
  const float* boxes  = (const float*)d_in[2];
  const float* w1     = (const float*)d_in[3];
  const float* b1     = (const float*)d_in[4];
  const float* g1     = (const float*)d_in[5];
  const float* bt1    = (const float*)d_in[6];
  const float* m1     = (const float*)d_in[7];
  const float* v1     = (const float*)d_in[8];
  const float* w2     = (const float*)d_in[9];
  const float* b2     = (const float*)d_in[10];
  const float* g2     = (const float*)d_in[11];
  const float* bt2    = (const float*)d_in[12];
  const float* m2     = (const float*)d_in[13];
  const float* v2     = (const float*)d_in[14];
  float* out = (float*)d_out;

  char* ws = (char*)d_ws;
  size_t off = 0;
  auto alloc = [&](size_t bytes) -> void* {
    void* p = ws + off;
    off = (off + bytes + 255) & ~(size_t)255;
    return p;
  };
  _Float16* roiA = (_Float16*)alloc((size_t)NROI * K1 * sizeof(_Float16));   // 102.8 MB
  _Float16* w1h  = (_Float16*)alloc((size_t)HID  * K1 * sizeof(_Float16));   //  12.8 MB
  _Float16* h1   = (_Float16*)alloc((size_t)NROI * HID * sizeof(_Float16));  //   4.2 MB
  _Float16* w2h  = (_Float16*)alloc((size_t)DOUT * HID * sizeof(_Float16));  //   0.5 MB

  roi_align_f16<<<dim3(NROI), dim3(256), 0, stream>>>(images, kfidx, boxes, roiA);

  const int nw1 = HID * K1;
  cvt_f32_f16<<<dim3((nw1 + 255) / 256), dim3(256), 0, stream>>>(w1, w1h, nw1);
  const int nw2 = DOUT * HID;
  cvt_f32_f16<<<dim3((nw2 + 255) / 256), dim3(256), 0, stream>>>(w2, w2h, nw2);

  // FC1 + BN1 + ReLU (M=4096, N=512, K=12544), f16 out for FC2
  gemm_bn_relu<0><<<dim3(NROI / 128, HID / 64), dim3(128), 0, stream>>>(
      roiA, w1h, K1, b1, g1, bt1, m1, v1, (void*)h1);

  // FC2 + BN2 + ReLU + max over 8 frames, writes nodes [512,512] directly
  gemm_bn_relu<1><<<dim3(NROI / 128, DOUT / 64), dim3(128), 0, stream>>>(
      h1, w2h, HID, b2, g2, bt2, m2, v2, (void*)out);
}
